// VideoMoELayer_8761733284172
// MI455X (gfx1250) — matrix operand order repaired and checked by reading, not compiled, not run on hardware
//
#include <hip/hip_runtime.h>
#include <stdint.h>

// Problem constants (reference: B,S,H,E,D,K = 2,4096,1024,8,2048,2)
constexpr int CB = 2, CS = 4096, CH = 1024, CE = 8, CD = 2048, CK = 2;
constexpr int CT = CB * CS;   // 8192 tokens
constexpr int TK = CT * CK;   // 16384 (expert,token) assignments

// GEMM tiling
constexpr int MT = 128, NT = 128, KC = 64, KP = 72; // KP pads LDS rows (bank-conflict break)

typedef __attribute__((ext_vector_type(16))) __bf16        v16bf;
typedef __attribute__((ext_vector_type(8)))  float         v8f;
typedef __attribute__((ext_vector_type(4)))  unsigned int  u32x4;
typedef __attribute__((ext_vector_type(2)))  unsigned int  u32x2;
typedef __attribute__((ext_vector_type(4)))  float         f32x4;

union FragBF { u32x4 q[2]; v16bf v; };

__device__ __forceinline__ unsigned short f2bf(float f) {
  unsigned int u = __float_as_uint(f);
  u += 0x7FFFu + ((u >> 16) & 1u);          // round-to-nearest-even
  return (unsigned short)(u >> 16);
}
__device__ __forceinline__ unsigned int f2bf2(float lo, float hi) {
  return (unsigned int)f2bf(lo) | ((unsigned int)f2bf(hi) << 16);
}
__device__ __forceinline__ float bf2f(unsigned short h) {
  return __uint_as_float(((unsigned int)h) << 16);
}
__device__ __forceinline__ float gelu_exact(float v) {
  return 0.5f * v * (1.0f + erff(v * 0.70710678118654752f));
}

// ---------------------------------------------------------------- init
__global__ void moe_init_kernel(int* counts, float* prob_sum) {
  if (threadIdx.x < CE) { counts[threadIdx.x] = 0; prob_sum[threadIdx.x] = 0.0f; }
}

// ---------------------------------------------------------------- router
// One wave32 per token: 8 dot products over H, softmax, top-2 (lowest-index ties),
// renormalized gates; block-aggregated aux-loss accumulators.
__global__ void __launch_bounds__(256)
moe_router_kernel(const float* __restrict__ x, const float* __restrict__ gw,
                  int* __restrict__ topk_idx, float* __restrict__ topk_p,
                  int* __restrict__ counts, float* __restrict__ prob_sum) {
  __shared__ float s_ps[CE];
  __shared__ int   s_cn[CE];
  if (threadIdx.x < CE) { s_ps[threadIdx.x] = 0.0f; s_cn[threadIdx.x] = 0; }
  __syncthreads();

  const int lane = threadIdx.x & 31;
  const int t    = blockIdx.x * 8 + (threadIdx.x >> 5);

  float acc[CE];
#pragma unroll
  for (int e = 0; e < CE; ++e) acc[e] = 0.0f;
  const float* xr = x + (size_t)t * CH;
  for (int h = lane; h < CH; h += 32) {
    float xv = xr[h];
#pragma unroll
    for (int e = 0; e < CE; ++e) acc[e] += xv * gw[e * CH + h];
  }
#pragma unroll
  for (int s = 16; s > 0; s >>= 1)
#pragma unroll
    for (int e = 0; e < CE; ++e) acc[e] += __shfl_xor(acc[e], s, 32);

  if (lane == 0) {
    float m = acc[0];
#pragma unroll
    for (int e = 1; e < CE; ++e) m = fmaxf(m, acc[e]);
    float p[CE], sum = 0.0f;
#pragma unroll
    for (int e = 0; e < CE; ++e) { p[e] = expf(acc[e] - m); sum += p[e]; }
    float inv = 1.0f / sum;
#pragma unroll
    for (int e = 0; e < CE; ++e) p[e] *= inv;

    int i0 = 0; float b0 = p[0];
#pragma unroll
    for (int e = 1; e < CE; ++e) if (p[e] > b0) { b0 = p[e]; i0 = e; }
    int i1 = -1; float b1v = -1.0f;
#pragma unroll
    for (int e = 0; e < CE; ++e) if (e != i0 && p[e] > b1v) { b1v = p[e]; i1 = e; }

    float gs = 1.0f / (b0 + b1v);
    topk_idx[t * 2 + 0] = i0; topk_idx[t * 2 + 1] = i1;
    topk_p[t * 2 + 0] = b0 * gs; topk_p[t * 2 + 1] = b1v * gs;
#pragma unroll
    for (int e = 0; e < CE; ++e) atomicAdd(&s_ps[e], p[e]);
    atomicAdd(&s_cn[i0], 1); atomicAdd(&s_cn[i1], 1);
  }
  __syncthreads();
  if (threadIdx.x < CE) {
    atomicAdd(&prob_sum[threadIdx.x], s_ps[threadIdx.x]);
    atomicAdd(&counts[threadIdx.x], s_cn[threadIdx.x]);
  }
}

// ---------------------------------------------------------------- prep (prefix + aux loss)
__global__ void moe_prep_kernel(const int* __restrict__ counts, int* __restrict__ offsets,
                                int* __restrict__ cursor, const float* __restrict__ prob_sum,
                                float* __restrict__ aux_out) {
  if (threadIdx.x < CE) cursor[threadIdx.x] = 0;
  if (threadIdx.x == 0) {
    int a = 0;
    for (int e = 0; e < CE; ++e) { offsets[e] = a; a += counts[e]; }
    // aux = E * sum_e (count_e / B) * (prob_sum_e / T)
    float s = 0.0f;
    for (int e = 0; e < CE; ++e)
      s += ((float)counts[e] / (float)CB) * (prob_sum[e] / (float)CT);
    aux_out[0] = (float)CE * s;
  }
}

// ---------------------------------------------------------------- scatter
__global__ void moe_scatter_kernel(const int* __restrict__ topk_idx,
                                   const int* __restrict__ offsets, int* __restrict__ cursor,
                                   int* __restrict__ token_list, int* __restrict__ slot_of) {
  int t = blockIdx.x * 256 + threadIdx.x;
  if (t >= CT) return;
#pragma unroll
  for (int k = 0; k < CK; ++k) {
    int e = topk_idx[t * 2 + k];
    int pos = offsets[e] + atomicAdd(&cursor[e], 1);
    token_list[pos] = t;
    slot_of[t * 2 + k] = pos;
  }
}

// ---------------------------------------------------------------- GEMM1: hid = gelu(x_g @ w1[e] + b1[e])
// Double-buffered LDS + register prefetch; one barrier per K-slab.
__global__ void __launch_bounds__(256)
moe_gemm1_kernel(const float* __restrict__ x, const float* __restrict__ w1,
                 const float* __restrict__ b1, const int* __restrict__ counts,
                 const int* __restrict__ offsets, const int* __restrict__ token_list,
                 unsigned short* __restrict__ hid) {
  const int e = blockIdx.z;
  const int cnt = counts[e];
  const int m0 = blockIdx.x * MT;
  if (m0 >= cnt) return;
  const int n0 = blockIdx.y * NT;
  const int off = offsets[e];

  __shared__ unsigned short sA[2][MT * KP];   // [row][k]  (tokens)
  __shared__ unsigned short sB[2][NT * KP];   // [col][k]  (K-contiguous per column)

  const int tid = threadIdx.x, lane = tid & 31, wid = tid >> 5;
  const int wm = wid >> 1, wn = wid & 1;      // 4x2 wave grid -> wave owns 32x64
  const int half = lane >> 4, l15 = lane & 15;

  // A staging geometry: thread covers rows (tid>>4)+16i, cols aCol..aCol+3 of the slab
  const int aRow = tid >> 4, aCol = (tid & 15) * 4;
  // B staging geometry: thread covers k-pair k2(i), n-quad bN
  const int bN = (tid & 31) * 4;

  // gather-row token ids (hoisted out of the K loop)
  int tokRow[8];
#pragma unroll
  for (int i = 0; i < 8; ++i) {
    int gm = m0 + aRow + 16 * i;
    tokRow[i] = token_list[off + ((gm < cnt) ? gm : 0)];
  }

  v8f acc[2][4];
#pragma unroll
  for (int i = 0; i < 2; ++i)
#pragma unroll
    for (int j = 0; j < 4; ++j)
#pragma unroll
      for (int r = 0; r < 8; ++r) acc[i][j][r] = 0.0f;

  f32x4 ra[8];
  f32x4 rb[4][2];

  auto loadA = [&](int kc0) {
#pragma unroll
    for (int i = 0; i < 8; ++i)
      ra[i] = *(const f32x4*)&x[(size_t)tokRow[i] * CH + kc0 + aCol];
  };
  auto loadB = [&](int kc0) {
#pragma unroll
    for (int i = 0; i < 4; ++i) {
      int k2 = ((tid >> 5) + 8 * i) * 2;
      rb[i][0] = *(const f32x4*)&w1[((size_t)e * CH + kc0 + k2 + 0) * CD + n0 + bN];
      rb[i][1] = *(const f32x4*)&w1[((size_t)e * CH + kc0 + k2 + 1) * CD + n0 + bN];
    }
  };
  auto stage = [&](int buf) {
#pragma unroll
    for (int i = 0; i < 8; ++i) {
      u32x2 p;
      p.x = f2bf2(ra[i].x, ra[i].y);
      p.y = f2bf2(ra[i].z, ra[i].w);
      *(u32x2*)&sA[buf][(aRow + 16 * i) * KP + aCol] = p;
    }
#pragma unroll
    for (int i = 0; i < 4; ++i) {
      int k2 = ((tid >> 5) + 8 * i) * 2;
      unsigned short* bp = &sB[buf][bN * KP + k2];
      *(unsigned int*)(bp + 0 * KP) = f2bf2(rb[i][0].x, rb[i][1].x);
      *(unsigned int*)(bp + 1 * KP) = f2bf2(rb[i][0].y, rb[i][1].y);
      *(unsigned int*)(bp + 2 * KP) = f2bf2(rb[i][0].z, rb[i][1].z);
      *(unsigned int*)(bp + 3 * KP) = f2bf2(rb[i][0].w, rb[i][1].w);
    }
  };

  constexpr int NS = CH / KC;  // 16
  loadA(0); loadB(0);
  int buf = 0;
  for (int s = 0; s < NS; ++s) {
    stage(buf);
    __syncthreads();
    if (s + 1 < NS) { loadA((s + 1) * KC); loadB((s + 1) * KC); }

#pragma unroll
    for (int ks = 0; ks < KC; ks += 32) {
      FragBF a[2], b[4];
#pragma unroll
      for (int i = 0; i < 2; ++i) {  // A 16x32: lane<16 K{0..7,16..23}, lane>=16 K{8..15,24..31}
        const unsigned short* p = &sA[buf][(wm * 32 + i * 16 + l15) * KP + ks + half * 8];
        a[i].q[0] = *(const u32x4*)(p);
        a[i].q[1] = *(const u32x4*)(p + 16);
      }
#pragma unroll
      for (int j = 0; j < 4; ++j) {  // B 32x16: lane holds col, lane<16 K{0..7,16..23}, lane>=16 K{8..15,24..31}
        const unsigned short* p = &sB[buf][(wn * 64 + j * 16 + l15) * KP + ks + half * 8];
        b[j].q[0] = *(const u32x4*)(p);
        b[j].q[1] = *(const u32x4*)(p + 16);
      }
#pragma unroll
      for (int i = 0; i < 2; ++i)
#pragma unroll
        for (int j = 0; j < 4; ++j)
          acc[i][j] = __builtin_amdgcn_wmma_f32_16x16x32_bf16(
              false, a[i].v, false, b[j].v, (short)0, acc[i][j], false, false);
    }
    buf ^= 1;
  }

  // epilogue: bias + exact GELU, bf16 store  (C/D: VGPR r -> row base+half*8+r, col = lane&15)
  const bool full = (m0 + MT <= cnt);
#pragma unroll
  for (int j = 0; j < 4; ++j) {
    int col = n0 + wn * 64 + j * 16 + l15;
    float bias = b1[(size_t)e * CD + col];
#pragma unroll
    for (int i = 0; i < 2; ++i) {
      int rbase = m0 + wm * 32 + i * 16 + half * 8;
      if (full || rbase + 7 < cnt) {
#pragma unroll
        for (int r = 0; r < 8; ++r)
          hid[(size_t)(off + rbase + r) * CD + col] = f2bf(gelu_exact(acc[i][j][r] + bias));
      } else {
#pragma unroll
        for (int r = 0; r < 8; ++r)
          if (rbase + r < cnt)
            hid[(size_t)(off + rbase + r) * CD + col] = f2bf(gelu_exact(acc[i][j][r] + bias));
      }
    }
  }
}

// ---------------------------------------------------------------- GEMM2: y = hid @ w2[e] + b2[e]
__global__ void __launch_bounds__(256)
moe_gemm2_kernel(const unsigned short* __restrict__ hid, const float* __restrict__ w2,
                 const float* __restrict__ b2, const int* __restrict__ counts,
                 const int* __restrict__ offsets, unsigned short* __restrict__ ybuf) {
  const int e = blockIdx.z;
  const int cnt = counts[e];
  const int m0 = blockIdx.x * MT;
  if (m0 >= cnt) return;
  const int n0 = blockIdx.y * NT;
  const int off = offsets[e];

  __shared__ unsigned short sA[2][MT * KP];
  __shared__ unsigned short sB[2][NT * KP];

  const int tid = threadIdx.x, lane = tid & 31, wid = tid >> 5;
  const int wm = wid >> 1, wn = wid & 1;
  const int half = lane >> 4, l15 = lane & 15;

  // A staging: thread covers rows (tid>>3)+32i, 8 bf16 at column c8
  const int aRow = tid >> 3, c8 = (tid & 7) * 8;
  const int bN = (tid & 31) * 4;

  int gRow[4];  // clamped slot rows (hoisted)
#pragma unroll
  for (int i = 0; i < 4; ++i) {
    int gm = m0 + aRow + 32 * i;
    gRow[i] = off + ((gm < cnt) ? gm : (cnt - 1));
  }

  v8f acc[2][4];
#pragma unroll
  for (int i = 0; i < 2; ++i)
#pragma unroll
    for (int j = 0; j < 4; ++j)
#pragma unroll
      for (int r = 0; r < 8; ++r) acc[i][j][r] = 0.0f;

  u32x4 ra[4];
  f32x4 rb[4][2];

  auto loadA = [&](int kc0) {
#pragma unroll
    for (int i = 0; i < 4; ++i)
      ra[i] = *(const u32x4*)&hid[(size_t)gRow[i] * CD + kc0 + c8];
  };
  auto loadB = [&](int kc0) {
#pragma unroll
    for (int i = 0; i < 4; ++i) {
      int k2 = ((tid >> 5) + 8 * i) * 2;
      rb[i][0] = *(const f32x4*)&w2[((size_t)e * CD + kc0 + k2 + 0) * CH + n0 + bN];
      rb[i][1] = *(const f32x4*)&w2[((size_t)e * CD + kc0 + k2 + 1) * CH + n0 + bN];
    }
  };
  auto stage = [&](int buf) {
#pragma unroll
    for (int i = 0; i < 4; ++i)
      *(u32x4*)&sA[buf][(aRow + 32 * i) * KP + c8] = ra[i];
#pragma unroll
    for (int i = 0; i < 4; ++i) {
      int k2 = ((tid >> 5) + 8 * i) * 2;
      unsigned short* bp = &sB[buf][bN * KP + k2];
      *(unsigned int*)(bp + 0 * KP) = f2bf2(rb[i][0].x, rb[i][1].x);
      *(unsigned int*)(bp + 1 * KP) = f2bf2(rb[i][0].y, rb[i][1].y);
      *(unsigned int*)(bp + 2 * KP) = f2bf2(rb[i][0].z, rb[i][1].z);
      *(unsigned int*)(bp + 3 * KP) = f2bf2(rb[i][0].w, rb[i][1].w);
    }
  };

  constexpr int NS = CD / KC;  // 32
  loadA(0); loadB(0);
  int buf = 0;
  for (int s = 0; s < NS; ++s) {
    stage(buf);
    __syncthreads();
    if (s + 1 < NS) { loadA((s + 1) * KC); loadB((s + 1) * KC); }

#pragma unroll
    for (int ks = 0; ks < KC; ks += 32) {
      FragBF a[2], b[4];
#pragma unroll
      for (int i = 0; i < 2; ++i) {
        const unsigned short* p = &sA[buf][(wm * 32 + i * 16 + l15) * KP + ks + half * 8];
        a[i].q[0] = *(const u32x4*)(p);
        a[i].q[1] = *(const u32x4*)(p + 16);
      }
#pragma unroll
      for (int j = 0; j < 4; ++j) {
        const unsigned short* p = &sB[buf][(wn * 64 + j * 16 + l15) * KP + ks + half * 8];
        b[j].q[0] = *(const u32x4*)(p);
        b[j].q[1] = *(const u32x4*)(p + 16);
      }
#pragma unroll
      for (int i = 0; i < 2; ++i)
#pragma unroll
        for (int j = 0; j < 4; ++j)
          acc[i][j] = __builtin_amdgcn_wmma_f32_16x16x32_bf16(
              false, a[i].v, false, b[j].v, (short)0, acc[i][j], false, false);
    }
    buf ^= 1;
  }

  const bool full = (m0 + MT <= cnt);
#pragma unroll
  for (int j = 0; j < 4; ++j) {
    int col = n0 + wn * 64 + j * 16 + l15;
    float bias = b2[(size_t)e * CH + col];
#pragma unroll
    for (int i = 0; i < 2; ++i) {
      int rbase = m0 + wm * 32 + i * 16 + half * 8;
      if (full || rbase + 7 < cnt) {
#pragma unroll
        for (int r = 0; r < 8; ++r)
          ybuf[(size_t)(off + rbase + r) * CH + col] = f2bf(acc[i][j][r] + bias);
      } else {
#pragma unroll
        for (int r = 0; r < 8; ++r)
          if (rbase + r < cnt)
            ybuf[(size_t)(off + rbase + r) * CH + col] = f2bf(acc[i][j][r] + bias);
      }
    }
  }
}

// ---------------------------------------------------------------- combine: out[t] = g0*y[s0] + g1*y[s1]
__global__ void __launch_bounds__(256)
moe_combine_kernel(const unsigned short* __restrict__ ybuf, const int* __restrict__ slot_of,
                   const float* __restrict__ topk_p, float* __restrict__ out) {
  int t = blockIdx.x;
  int s0 = slot_of[t * 2 + 0], s1 = slot_of[t * 2 + 1];
  float g0 = topk_p[t * 2 + 0], g1 = topk_p[t * 2 + 1];
  int h = threadIdx.x * 4;  // 256 threads x 4 cols = CH
  u32x2 ya = *(const u32x2*)&ybuf[(size_t)s0 * CH + h];
  u32x2 yb = *(const u32x2*)&ybuf[(size_t)s1 * CH + h];
  f32x4 o;
  o.x = g0 * bf2f((unsigned short)(ya.x & 0xFFFF)) + g1 * bf2f((unsigned short)(yb.x & 0xFFFF));
  o.y = g0 * bf2f((unsigned short)(ya.x >> 16))    + g1 * bf2f((unsigned short)(yb.x >> 16));
  o.z = g0 * bf2f((unsigned short)(ya.y & 0xFFFF)) + g1 * bf2f((unsigned short)(yb.y & 0xFFFF));
  o.w = g0 * bf2f((unsigned short)(ya.y >> 16))    + g1 * bf2f((unsigned short)(yb.y >> 16));
  *(f32x4*)&out[(size_t)t * CH + h] = o;
}

// ---------------------------------------------------------------- launch
extern "C" void kernel_launch(void* const* d_in, const int* in_sizes, int n_in,
                              void* d_out, int out_size, void* d_ws, size_t ws_size,
                              hipStream_t stream) {
  (void)in_sizes; (void)n_in; (void)out_size; (void)ws_size;
  const float* x  = (const float*)d_in[0];
  const float* gw = (const float*)d_in[1];
  const float* w1 = (const float*)d_in[2];
  const float* b1 = (const float*)d_in[3];
  const float* w2 = (const float*)d_in[4];
  const float* b2 = (const float*)d_in[5];
  float* out = (float*)d_out;

  char* ws = (char*)d_ws;
  size_t o = 0;
  auto take = [&](size_t bytes) {
    char* p = ws + o;
    o += bytes;
    o = (o + 255) & ~(size_t)255;
    return p;
  };
  int*   topk_idx   = (int*)  take((size_t)CT * 2 * sizeof(int));
  float* topk_p     = (float*)take((size_t)CT * 2 * sizeof(float));
  int*   counts     = (int*)  take(CE * sizeof(int));
  int*   offsets    = (int*)  take(CE * sizeof(int));
  int*   cursor     = (int*)  take(CE * sizeof(int));
  float* prob_sum   = (float*)take(CE * sizeof(float));
  int*   token_list = (int*)  take((size_t)TK * sizeof(int));
  int*   slot_of    = (int*)  take((size_t)TK * sizeof(int));
  unsigned short* hid  = (unsigned short*)take((size_t)TK * CD * sizeof(unsigned short));
  unsigned short* ybuf = (unsigned short*)take((size_t)TK * CH * sizeof(unsigned short));

  moe_init_kernel<<<1, 32, 0, stream>>>(counts, prob_sum);
  moe_router_kernel<<<CT / 8, 256, 0, stream>>>(x, gw, topk_idx, topk_p, counts, prob_sum);
  moe_prep_kernel<<<1, 32, 0, stream>>>(counts, offsets, cursor, prob_sum,
                                        out + (size_t)CT * CH);
  moe_scatter_kernel<<<CT / 256, 256, 0, stream>>>(topk_idx, offsets, cursor, token_list,
                                                   slot_of);
  dim3 g1(CT / MT, CD / NT, CE);  // 64 x 16 x 8, blocks beyond expert count early-exit
  moe_gemm1_kernel<<<g1, 256, 0, stream>>>(x, w1, b1, counts, offsets, token_list, hid);
  dim3 g2(CT / MT, CH / NT, CE);  // 64 x 8 x 8
  moe_gemm2_kernel<<<g2, 256, 0, stream>>>(hid, w2, b2, counts, offsets, ybuf);
  moe_combine_kernel<<<CT, 256, 0, stream>>>(ybuf, slot_of, topk_p, out);
}
